// iEFLModule_51797305590115
// MI455X (gfx1250) — matrix operand order repaired and checked by reading, not compiled, not run on hardware
//
#include <hip/hip_runtime.h>
#include <hip/hip_bf16.h>
#include <math.h>

// ---------------- problem constants ----------------
#define E_       256
#define HW_      4096
#define BS_      32
#define ROWS_    (HW_ * BS_)     // 131072 memory rows
#define NQ_      27              // NOBJ*K*K
#define NB_      864             // NQ_*BS_
#define NPAD_    896             // padded to 14*64
#define FFN_     2048
#define STEPS_   3

typedef __attribute__((ext_vector_type(16))) __bf16 v16bf;
typedef __attribute__((ext_vector_type(8)))  float  v8f;

union Frag  { uint4 q[2]; v16bf v; };
union Pack8 { uint4 q; unsigned short h[8]; };

__device__ __forceinline__ unsigned short f2bf(float f) {
  union { float f; unsigned u; } x; x.f = f;
  unsigned r = x.u + 0x7fffu + ((x.u >> 16) & 1u);   // round-to-nearest-even
  return (unsigned short)(r >> 16);
}
__device__ __forceinline__ float bf2f(unsigned short h) {
  union { unsigned u; float f; } x; x.u = ((unsigned)h) << 16;
  return x.f;
}

// =====================================================================
// Phase A: L = (mem+pos)@Wk + bk ; V = (mem+pos)@Wv + bv   (bf16 WMMA, f32 acc)
// Block = 64 rows (32 hw x 2 batch) x all 256 cols, 8 waves.
// Double-buffered weight slabs (aliasing the dead f32 transpose buffer),
// single barrier per K-slab; B-fragment register loads pipelined one tile
// ahead so WMMA overlaps ds_load latency (partial s_wait_dscnt).
// Output layout: [col][blk*64 + localRow] (row order irrelevant: all
// consumers are order-invariant per-column reductions).
// =====================================================================
__global__ __launch_bounds__(256) void k_phaseA_gemm(
    const float* __restrict__ fe, const float* __restrict__ pos,
    const float* __restrict__ wk, const float* __restrict__ wv,
    const float* __restrict__ bk, const float* __restrict__ bv,
    unsigned short* __restrict__ Lb, unsigned short* __restrict__ Vb)
{
  extern __shared__ __align__(16) char smem[];
  unsigned short* Abf = (unsigned short*)smem;                  // [64][264] bf16
  float*          X   = (float*)(smem + 64 * 264 * 2);          // [64][256] f32 (prologue only)
  unsigned short* Wl0 = (unsigned short*)(smem + 64 * 264 * 2); // [2][16][32][16] (aliases X)
  unsigned short* Wl1 = Wl0 + 2 * 16 * 32 * 16;                 // second buffer

  const int tid = threadIdx.x;
  const int blk = blockIdx.x;
  const int hw0 = (blk & 127) * 32;
  const int b0  = (blk >> 7) * 2;

  // stage f_e tile: contiguous 128B runs along hw (f_e is [b][e][hw])
  for (int c = tid; c < 512; c += 256) {
    const int bl = c >> 8, e = c & 255;
    const float4* src = (const float4*)(fe + ((size_t)(b0 + bl) * E_ + e) * HW_ + hw0);
    #pragma unroll
    for (int j = 0; j < 8; ++j) {
      float4 v = src[j];
      X[((j * 4 + 0) * 2 + bl) * E_ + e] = v.x;
      X[((j * 4 + 1) * 2 + bl) * E_ + e] = v.y;
      X[((j * 4 + 2) * 2 + bl) * E_ + e] = v.z;
      X[((j * 4 + 3) * 2 + bl) * E_ + e] = v.w;
    }
  }
  __syncthreads();
  // A = bf16(f_e + pos); pos rows are e-contiguous (row r = hw*32+b)
  for (int i = tid; i < 64 * E_; i += 256) {
    const int lr = i >> 8, e = i & 255;
    const size_t rg = (size_t)(hw0 + (lr >> 1)) * BS_ + b0 + (lr & 1);
    Abf[lr * 264 + e] = f2bf(X[lr * E_ + e] + pos[rg * E_ + e]);
  }
  __syncthreads();   // X dead from here; Wl0/Wl1 take over the region

  const int wave = tid >> 5, lane = tid & 31;
  const int mtile = wave >> 1;       // 4 M-tiles of 16 rows
  const int mat   = wave & 1;        // 0 -> Wk/L, 1 -> Wv/V

  // stage swizzled weight slab ks into dst (both matrices), f32->bf16 inline.
  auto stageW = [&](int ks, unsigned short* dst) {
    const int k0 = ks * 32;
    for (int i = tid; i < 2 * 32 * E_; i += 256) {
      const int m = i >> 13, kk = (i >> 8) & 31, n = i & 255;
      const float w = (m ? wv : wk)[(size_t)(k0 + kk) * E_ + n];
      const int ln = (n & 15) + ((kk >> 4) << 4), ii = kk & 15, nt = n >> 4;
      dst[((m * 16 + nt) * 32 + ln) * 16 + ii] = f2bf(w);
    }
  };

  v8f acc[16];
  #pragma unroll
  for (int nt = 0; nt < 16; ++nt)
    #pragma unroll
    for (int v = 0; v < 8; ++v) acc[nt][v] = 0.f;

  stageW(0, Wl0);
  for (int ks = 0; ks < 8; ++ks) {
    __syncthreads();                               // slab ks staged; prev compute done
    if (ks < 7) stageW(ks + 1, (ks & 1) ? Wl0 : Wl1);
    const unsigned short* Wcur = (ks & 1) ? Wl1 : Wl0;
    const unsigned short* wb = Wcur + (mat * 16) * 32 * 16;
    const int k0 = ks * 32;

    Frag af;
    {
      const int row = mtile * 16 + (lane & 15);
      const int kb  = (lane >> 4) << 4;            // contiguous A layout: hi-lanes K+16
      af.q[0] = *(const uint4*)(Abf + row * 264 + k0 + kb);
      af.q[1] = *(const uint4*)(Abf + row * 264 + k0 + 8 + kb);
    }
    Frag bfr[2];
    {
      const uint4* bp = (const uint4*)(wb + (0 * 32 + lane) * 16);
      bfr[0].q[0] = bp[0]; bfr[0].q[1] = bp[1];
    }
    #pragma unroll
    for (int nt = 0; nt < 16; ++nt) {
      if (nt < 15) {                               // pipeline next B-fragment
        const uint4* bp = (const uint4*)(wb + ((nt + 1) * 32 + lane) * 16);
        bfr[(nt + 1) & 1].q[0] = bp[0];
        bfr[(nt + 1) & 1].q[1] = bp[1];
      }
      acc[nt] = __builtin_amdgcn_wmma_f32_16x16x32_bf16(false, af.v, false, bfr[nt & 1].v,
                                                        (short)0, acc[nt], false, false);
    }
  }

  // +bias, pack bf16, 16B stores (8 consecutive rows per lane per tile)
  unsigned short* dst = mat ? Vb : Lb;
  const float* bias   = mat ? bv : bk;
  const int rowb = mtile * 16 + ((lane >> 4) << 3);
  #pragma unroll
  for (int nt = 0; nt < 16; ++nt) {
    const int col = nt * 16 + (lane & 15);
    const float bb = bias[col];
    Pack8 pk;
    #pragma unroll
    for (int v = 0; v < 8; ++v) pk.h[v] = f2bf(acc[nt][v] + bb);
    *(uint4*)(dst + (size_t)col * ROWS_ + (size_t)blk * 64 + rowb) = pk.q;
  }
}

// ---------------- per-column max over 131072 rows (1 block / column) ----
__global__ __launch_bounds__(256) void k_colmax(const unsigned short* __restrict__ Lb,
                                                float* __restrict__ colmax)
{
  const int col = blockIdx.x, tid = threadIdx.x;
  const uint4* p = (const uint4*)(Lb + (size_t)col * ROWS_);
  float m = -3.4e38f;
  for (int i = tid; i < ROWS_ / 8; i += 256) {
    Pack8 pk; pk.q = p[i];
    #pragma unroll
    for (int j = 0; j < 8; ++j) m = fmaxf(m, bf2f(pk.h[j]));
  }
  #pragma unroll
  for (int o = 16; o > 0; o >>= 1) m = fmaxf(m, __shfl_xor(m, o, 32));
  __shared__ float sm[8];
  if ((tid & 31) == 0) sm[tid >> 5] = m;
  __syncthreads();
  if (tid == 0) {
    float r = sm[0];
    for (int w = 1; w < 8; ++w) r = fmaxf(r, sm[w]);
    colmax[col] = r;
  }
}

// ------- context[c] = sum(exp(L-mx)*V)/sum(exp(L-mx)) (1 block/column) ---
__global__ __launch_bounds__(256) void k_context(const unsigned short* __restrict__ Lb,
    const unsigned short* __restrict__ Vb, const float* __restrict__ colmax,
    float* __restrict__ ctx)
{
  const int col = blockIdx.x, tid = threadIdx.x;
  const float mx = colmax[col];
  const uint4* pl = (const uint4*)(Lb + (size_t)col * ROWS_);
  const uint4* pv = (const uint4*)(Vb + (size_t)col * ROWS_);
  float num = 0.f, den = 0.f;
  for (int i = tid; i < ROWS_ / 8; i += 256) {
    Pack8 a, b; a.q = pl[i]; b.q = pv[i];
    #pragma unroll
    for (int j = 0; j < 8; ++j) {
      const float e = __expf(bf2f(a.h[j]) - mx);
      den += e; num += e * bf2f(b.h[j]);
    }
  }
  #pragma unroll
  for (int o = 16; o > 0; o >>= 1) { num += __shfl_xor(num, o, 32); den += __shfl_xor(den, o, 32); }
  __shared__ float sn[8], sd[8];
  if ((tid & 31) == 0) { sn[tid >> 5] = num; sd[tid >> 5] = den; }
  __syncthreads();
  if (tid == 0) {
    float n = 0.f, d = 0.f;
    for (int w = 0; w < 8; ++w) { n += sn[w]; d += sd[w]; }
    ctx[col] = n / d;
  }
}

// ---------------- Phase B init: query-pos encodings + F0 = shape_emb ----
__global__ __launch_bounds__(256) void k_binit(const float* __restrict__ shape_map,
    float* __restrict__ F, float* __restrict__ qpos)
{
  const int gt = blockIdx.x * 256 + threadIdx.x, stride = gridDim.x * 256;
  for (int i = gt; i < 9 * E_; i += stride) {
    const int p = i >> 8, e = i & 255;
    const int half = e >> 7, ie = e & 127;
    const float P = half ? (float)(p % 3 + 1) : (float)(p / 3 + 1);  // x+1 / y+1
    const float expo = (float)((ie >> 1) << 1) / 128.f;
    const float arg = P / __powf(10000.f, expo);
    qpos[i] = (ie & 1) ? __cosf(arg) : __sinf(arg);
  }
  for (int i = gt; i < NPAD_ * E_; i += stride) {
    const int r = i >> 8, e = i & 255, nq = r >> 5;
    F[i] = (nq < NQ_) ? shape_map[nq * E_ + e] : 0.f;
  }
}

// -------- k/v projections of shape_emb (batch-invariant: only 27x256) ----
__global__ __launch_bounds__(256) void k_kv(const float* __restrict__ shape_map,
    const float* __restrict__ wkm, const float* __restrict__ bkm,
    const float* __restrict__ wvm, const float* __restrict__ bvm,
    float* __restrict__ kp, float* __restrict__ vp)
{
  const int g = blockIdx.x * 256 + threadIdx.x;
  if (g >= 2 * NQ_ * E_) return;
  const int m = g / (NQ_ * E_), rem = g % (NQ_ * E_), s = rem >> 8, e = rem & 255;
  const float* w = m ? wvm : wkm;
  float acc = m ? bvm[e] : bkm[e];
  for (int k = 0; k < E_; ++k) acc += shape_map[s * E_ + k] * w[k * E_ + e];
  (m ? vp : kp)[s * E_ + e] = acc;
}

// ---------------- MHA attention core: one block per (l,b) row ------------
__global__ __launch_bounds__(256) void k_attn(const float* __restrict__ qp,
    const float* __restrict__ kp, const float* __restrict__ vp, float* __restrict__ o)
{
  __shared__ float skp[NQ_ * E_], svp[NQ_ * E_], sq[E_];
  const int row = blockIdx.x, tid = threadIdx.x;
  for (int i = tid; i < NQ_ * E_; i += 256) { skp[i] = kp[i]; svp[i] = vp[i]; }
  sq[tid] = qp[(size_t)row * E_ + tid];
  __syncthreads();
  const int h = tid >> 5, lane = tid & 31;   // wave = head (Dh = 32 = wave32!)
  float score = -3.4e38f;
  if (lane < NQ_) {
    float d = 0.f;
    for (int k = 0; k < 32; ++k) d += sq[h * 32 + k] * skp[lane * E_ + h * 32 + k];
    score = d * 0.17677669529663687f;        // 1/sqrt(32)
  }
  float m = score;
  #pragma unroll
  for (int off = 16; off > 0; off >>= 1) m = fmaxf(m, __shfl_xor(m, off, 32));
  const float e = (lane < NQ_) ? __expf(score - m) : 0.f;
  float s = e;
  #pragma unroll
  for (int off = 16; off > 0; off >>= 1) s += __shfl_xor(s, off, 32);
  const float att = e / s;
  float acc = 0.f;
  for (int si = 0; si < NQ_; ++si)
    acc += __shfl(att, si, 32) * svp[si * E_ + h * 32 + lane];
  o[(size_t)row * E_ + h * 32 + lane] = acc;
}

// =====================================================================
// Generic Phase-B WMMA GEMM: C = A[896,K] @ W[K,Ncols] + bias, fused:
//  mode 0: (F+qpos)@Wq      -> qp
//  mode 1: o@Wo; F=LN1(F+C)
//  mode 2: (F+qpos)@Wla; row-softmax(*E^-0.5); la=q*ctx; F=LN2(F+la)
//  mode 3: F@W1; gelu -> H    (col window cb)
//  mode 4: H@W2; F=LN3(F+C); emit to d_out
// Double-buffered A/W slabs (1 barrier per slab), pipelined B-fragments.
// Block owns 64 full rows x 256-col window -> row-wise epilogues local.
// =====================================================================
__global__ __launch_bounds__(256) void k_gemmB(
    const float* __restrict__ Ain, int lda,
    const float* __restrict__ W, int ldw,
    const float* __restrict__ bias,
    const float* __restrict__ addend,     // qpos (mode 0/2) or null
    float* __restrict__ F,
    float* __restrict__ aux,              // qp (0) / H (3)
    const float* __restrict__ ctx,
    const float* __restrict__ ln_g, const float* __restrict__ ln_b,
    float* __restrict__ dout, int mode)
{
  extern __shared__ __align__(16) char smem[];
  unsigned short* Asl0 = (unsigned short*)smem;             // [64][48] bf16
  unsigned short* Asl1 = (unsigned short*)(smem + 6144);
  unsigned short* Wl0  = (unsigned short*)(smem + 12288);   // [16][32][16] swizzled
  unsigned short* Wl1  = (unsigned short*)(smem + 28672);
  float*          Cbuf = (float*)(smem + 45056);            // [64][256]
  const int tid = threadIdx.x;
  const int rb = blockIdx.x, cb = blockIdx.y;
  const int wave = tid >> 5, lane = tid & 31;
  const int mtile = wave & 3, nt0 = (wave >> 2) * 8;

  auto stageA = [&](int ks, unsigned short* dst) {
    for (int i = tid; i < 64 * 32; i += 256) {
      const int r = i >> 5, kk = i & 31, kg = ks * 32 + kk;
      float v = Ain[(size_t)(rb * 64 + r) * lda + kg];
      if (addend) v += addend[(((rb * 64 + r) >> 5) % 9) * E_ + kg];
      dst[r * 48 + kk] = f2bf(v);
    }
  };
  auto stageW = [&](int ks, unsigned short* dst) {
    for (int i = tid; i < 32 * 256; i += 256) {
      const int kk = i >> 8, n = i & 255;
      const float w = W[(size_t)(ks * 32 + kk) * ldw + cb * 256 + n];
      const int ln = (n & 15) + ((kk >> 4) << 4), ii = kk & 15, nt = n >> 4;
      dst[(nt * 32 + ln) * 16 + ii] = f2bf(w);
    }
  };

  v8f acc[8];
  #pragma unroll
  for (int j = 0; j < 8; ++j)
    #pragma unroll
    for (int v = 0; v < 8; ++v) acc[j][v] = 0.f;

  const int nslab = lda >> 5;
  stageA(0, Asl0);
  stageW(0, Wl0);
  for (int ks = 0; ks < nslab; ++ks) {
    __syncthreads();                                   // slab ks staged; prev compute done
    if (ks + 1 < nslab) {
      stageA(ks + 1, (ks & 1) ? Asl0 : Asl1);
      stageW(ks + 1, (ks & 1) ? Wl0 : Wl1);
    }
    const unsigned short* Acur = (ks & 1) ? Asl1 : Asl0;
    const unsigned short* Wcur = (ks & 1) ? Wl1 : Wl0;

    Frag af;
    {
      const int row = mtile * 16 + (lane & 15);
      const int kb  = (lane >> 4) << 4;
      af.q[0] = *(const uint4*)(Acur + row * 48 + kb);
      af.q[1] = *(const uint4*)(Acur + row * 48 + 8 + kb);
    }
    Frag bfr[2];
    {
      const uint4* bp = (const uint4*)(Wcur + (nt0 * 32 + lane) * 16);
      bfr[0].q[0] = bp[0]; bfr[0].q[1] = bp[1];
    }
    #pragma unroll
    for (int j = 0; j < 8; ++j) {
      if (j < 7) {                                     // pipeline next B-fragment
        const uint4* bp = (const uint4*)(Wcur + ((nt0 + j + 1) * 32 + lane) * 16);
        bfr[(j + 1) & 1].q[0] = bp[0];
        bfr[(j + 1) & 1].q[1] = bp[1];
      }
      acc[j] = __builtin_amdgcn_wmma_f32_16x16x32_bf16(false, af.v, false, bfr[j & 1].v,
                                                       (short)0, acc[j], false, false);
    }
  }
  __syncthreads();

  // spill accumulators (+bias) to LDS C-buffer
  {
    const int row0 = mtile * 16 + ((lane >> 4) << 3);
    #pragma unroll
    for (int j = 0; j < 8; ++j) {
      const int col = (nt0 + j) * 16 + (lane & 15);
      const float bb = bias[cb * 256 + col];
      #pragma unroll
      for (int v = 0; v < 8; ++v) Cbuf[(row0 + v) * 256 + col] = acc[j][v] + bb;
    }
  }
  __syncthreads();

  if (mode == 0) {
    for (int i = tid; i < 64 * 256; i += 256) {
      const int r = i >> 8, c = i & 255;
      aux[(size_t)(rb * 64 + r) * ldw + cb * 256 + c] = Cbuf[i];
    }
  } else if (mode == 3) {
    for (int i = tid; i < 64 * 256; i += 256) {
      const int r = i >> 8, c = i & 255;
      const float v = Cbuf[i];
      aux[(size_t)(rb * 64 + r) * ldw + cb * 256 + c] =
          0.5f * v * (1.f + erff(v * 0.70710678118654752f));  // exact gelu
    }
  } else if (tid < 64) {   // row-wise epilogues: thread = row
    const int r = tid, rg = rb * 64 + r;
    if (mode == 2) {
      float mx = -3.4e38f;
      for (int e = 0; e < E_; ++e) mx = fmaxf(mx, Cbuf[r * E_ + e] * 0.0625f);
      float s = 0.f;
      for (int e = 0; e < E_; ++e) {
        const float ev = __expf(Cbuf[r * E_ + e] * 0.0625f - mx);
        Cbuf[r * E_ + e] = ev; s += ev;
      }
      const float inv = 1.f / s;
      for (int e = 0; e < E_; ++e)
        Cbuf[r * E_ + e] = F[(size_t)rg * E_ + e] + Cbuf[r * E_ + e] * inv * ctx[e];
    } else {  // mode 1 / 4: residual add
      for (int e = 0; e < E_; ++e) Cbuf[r * E_ + e] += F[(size_t)rg * E_ + e];
    }
    float s = 0.f, s2 = 0.f;
    for (int e = 0; e < E_; ++e) { const float v = Cbuf[r * E_ + e]; s += v; s2 += v * v; }
    const float mean = s * (1.f / 256.f);
    const float var  = s2 * (1.f / 256.f) - mean * mean;
    const float inv  = rsqrtf(var + 1e-5f);
    for (int e = 0; e < E_; ++e) {
      const float y = (Cbuf[r * E_ + e] - mean) * inv * ln_g[e] + ln_b[e];
      F[(size_t)rg * E_ + e] = y;
      if (mode == 4 && rg < NB_) dout[(size_t)rg * E_ + e] = y;
    }
  }
}

// =====================================================================
extern "C" void kernel_launch(void* const* d_in, const int* in_sizes, int n_in,
                              void* d_out, int out_size, void* d_ws, size_t ws_size,
                              hipStream_t stream)
{
  const float* fe   = (const float*)d_in[0];
  const float* pos  = (const float*)d_in[1];
  const float* shape= (const float*)d_in[3];
  const float* mwq  = (const float*)d_in[4];  const float* mbq = (const float*)d_in[5];
  const float* mwk  = (const float*)d_in[6];  const float* mbk = (const float*)d_in[7];
  const float* mwv  = (const float*)d_in[8];  const float* mbv = (const float*)d_in[9];
  const float* mwo  = (const float*)d_in[10]; const float* mbo = (const float*)d_in[11];
  const float* lwq  = (const float*)d_in[12]; const float* lbq = (const float*)d_in[13];
  const float* lwk  = (const float*)d_in[14]; const float* lbk = (const float*)d_in[15];
  const float* lwv  = (const float*)d_in[16]; const float* lbv = (const float*)d_in[17];
  const float* fw1  = (const float*)d_in[18]; const float* fb1 = (const float*)d_in[19];
  const float* fw2  = (const float*)d_in[20]; const float* fb2 = (const float*)d_in[21];
  const float* n1g  = (const float*)d_in[22]; const float* n1b = (const float*)d_in[23];
  const float* n2g  = (const float*)d_in[24]; const float* n2b = (const float*)d_in[25];
  const float* n3g  = (const float*)d_in[26]; const float* n3b = (const float*)d_in[27];
  float* out = (float*)d_out;
  (void)in_sizes; (void)n_in; (void)out_size; (void)ws_size;

  char* ws = (char*)d_ws;
  size_t off = 0;
  auto carve = [&](size_t bytes) {
    char* p = ws + off;
    off = (off + bytes + 255) & ~(size_t)255;
    return p;
  };
  unsigned short* Lb = (unsigned short*)carve((size_t)ROWS_ * E_ * 2);  // 64 MB
  unsigned short* Vb = (unsigned short*)carve((size_t)ROWS_ * E_ * 2);  // 64 MB
  float* colmax = (float*)carve(E_ * 4);
  float* ctx    = (float*)carve(E_ * 4);
  float* F      = (float*)carve((size_t)NPAD_ * E_ * 4);
  float* qp     = (float*)carve((size_t)NPAD_ * E_ * 4);
  float* o      = (float*)carve((size_t)NPAD_ * E_ * 4);
  float* H      = (float*)carve((size_t)NPAD_ * FFN_ * 4);
  float* kp     = (float*)carve(32 * E_ * 4);
  float* vp     = (float*)carve(32 * E_ * 4);
  float* qpos   = (float*)carve(16 * E_ * 4);

  const size_t smemA = 64 * 264 * 2 + 2 * (2 * 16 * 32 * 16) * 2;  // 99,328 B
  const size_t smemB = 45056 + 64 * 256 * 4;                       // 110,592 B

  // Phase A: dual GEMM -> bf16 L/V, then column softmax reduction -> context
  k_phaseA_gemm<<<2048, 256, smemA, stream>>>(fe, pos, lwk, lwv, lbk, lbv, Lb, Vb);
  k_colmax<<<E_, 256, 0, stream>>>(Lb, colmax);
  k_context<<<E_, 256, 0, stream>>>(Lb, Vb, colmax, ctx);

  // Phase B setup
  k_binit<<<64, 256, 0, stream>>>(shape, F, qpos);
  k_kv<<<(2 * NQ_ * E_ + 255) / 256, 256, 0, stream>>>(shape, mwk, mbk, mwv, mbv, kp, vp);

  for (int step = 0; step < STEPS_; ++step) {
    // qp = (F + qpos) @ Wq + bq
    k_gemmB<<<dim3(14, 1), 256, smemB, stream>>>(F, E_, mwq, E_, mbq, qpos, F, qp,
                                                 nullptr, nullptr, nullptr, nullptr, 0);
    // MHA core
    k_attn<<<NB_, 256, 0, stream>>>(qp, kp, vp, o);
    // F = LN1(F + o @ Wo + bo)
    k_gemmB<<<dim3(14, 1), 256, smemB, stream>>>(o, E_, mwo, E_, mbo, nullptr, F, nullptr,
                                                 nullptr, n1g, n1b, nullptr, 1);
    // linear attention: F = LN2(F + softmax((F+qpos)@Wla*scale) * context)
    k_gemmB<<<dim3(14, 1), 256, smemB, stream>>>(F, E_, lwq, E_, lbq, qpos, F, nullptr,
                                                 ctx, n2g, n2b, nullptr, 2);
    // H = gelu(F @ W1 + b1)
    k_gemmB<<<dim3(14, 8), 256, smemB, stream>>>(F, E_, fw1, FFN_, fb1, nullptr, F, H,
                                                 nullptr, nullptr, nullptr, nullptr, 3);
    // F = LN3(F + H @ W2 + b2); emit protos[step]
    k_gemmB<<<dim3(14, 1), 256, smemB, stream>>>(H, FFN_, fw2, E_, fb2, nullptr, F, nullptr,
                                                 nullptr, n3g, n3b,
                                                 out + (size_t)step * NB_ * E_, 4);
  }
}
